// AEModel_21938692948225
// MI455X (gfx1250) — matrix operand order repaired and checked by reading, not compiled, not run on hardware
//
#include <hip/hip_runtime.h>
#include <math.h>

#define NN 10000          // nodes
#define NE 320000         // edges
#define NR 32             // relations
#define DIM 256           // IN == H == OUT
#define RELIN 768
#define MAX_TILES (NE/16 + NR)

typedef __attribute__((ext_vector_type(16))) __bf16 v16bf;
typedef __attribute__((ext_vector_type(8)))  float  v8f;

struct __align__(16) U4 { unsigned x, y, z, w; };
struct __align__(32) U8 { U4 lo, hi; };

__device__ __forceinline__ unsigned short f2bf(float f) {
    union { float f; unsigned u; } c; c.f = f;
    unsigned u = c.u;
    u += 0x7FFFu + ((u >> 16) & 1u);   // round-to-nearest-even
    return (unsigned short)(u >> 16);
}

// ---------------- utility fills ----------------
__global__ __launch_bounds__(256) void k_fill_u32(unsigned* p, int n, unsigned v) {
    int i = blockIdx.x * 256 + threadIdx.x;
    if (i < n) p[i] = v;
}

// ---------------- edge bucketing by relation ----------------
__global__ __launch_bounds__(256) void k_hist(const int* __restrict__ et, int* cnt) {
    int e = blockIdx.x * 256 + threadIdx.x;
    if (e < NE) atomicAdd(&cnt[et[e]], 1);
}

__global__ void k_scan(int* ibuf) {  // single thread: 32 relations
    int* cnt = ibuf; int* off = ibuf + 32; int* cur = ibuf + 65; int* toff = ibuf + 97;
    off[0] = 0; toff[0] = 0;
    for (int r = 0; r < NR; r++) {
        off[r + 1]  = off[r] + cnt[r];
        toff[r + 1] = toff[r] + (cnt[r] + 15) / 16;
        cur[r] = off[r];
    }
}

__global__ __launch_bounds__(256) void k_desc(const int* __restrict__ ibuf, int2* desc) {
    int t = blockIdx.x * 256 + threadIdx.x;
    if (t >= MAX_TILES) return;
    const int* off = ibuf + 32; const int* toff = ibuf + 97;
    if (t >= toff[NR]) { desc[t] = make_int2(-1, 0); return; }
    int r = 0;
    while (r < NR - 1 && t >= toff[r + 1]) r++;
    int start = off[r] + (t - toff[r]) * 16;
    int c = off[r + 1] - start; if (c > 16) c = 16;
    desc[t] = make_int2(start, r | (c << 8));
}

__global__ __launch_bounds__(256) void k_sort(const int* __restrict__ src, const int* __restrict__ dst,
                                              const int* __restrict__ et, int* cur,
                                              int* srcS, int* dstS, int* eord) {
    int e = blockIdx.x * 256 + threadIdx.x;
    if (e >= NE) return;
    int r = et[e];
    int p = atomicAdd(&cur[r], 1);
    srcS[p] = src[e]; dstS[p] = dst[e]; eord[p] = e;
}

// ---------------- pack W[r] (K x N, row-major) into B-fragment layout ----------------
// fragment addr = (((r*8+kk)*16+nt)*32 + lane)*16 + i ; element = W[r][K][N],
// K = kk*32 + (i>>3)*16 + (lane>>4)*8 + (i&7) ,  N = nt*16 + (lane&15)
__global__ __launch_bounds__(256) void k_pack(const float* __restrict__ W, unsigned short* Wp, int rels) {
    int idx = blockIdx.x * 256 + threadIdx.x;
    if (idx >= rels * 65536) return;
    int i    = idx & 15;
    int lane = (idx >> 4) & 31;
    int nt   = (idx >> 9) & 15;
    int kk   = (idx >> 13) & 7;
    int r    = idx >> 16;
    int K    = kk * 32 + ((i >> 3) << 4) + ((lane >> 4) << 3) + (i & 7);
    int Ncol = nt * 16 + (lane & 15);
    Wp[idx] = f2bf(W[(r * DIM + K) * DIM + Ncol]);
}

// ---------------- qW[r] = W[r] @ q[r], kW[r] = W[r] @ k[r] ----------------
__global__ __launch_bounds__(256) void k_prep_qk(const float* __restrict__ W, const float* __restrict__ q,
                                                 const float* __restrict__ k, float* qW, float* kW) {
    int idx = blockIdx.x * 256 + threadIdx.x;
    if (idx >= NR * DIM) return;
    int r = idx >> 8, i = idx & 255;
    float aq = 0.f, ak = 0.f;
    for (int o = 0; o < DIM; o++) {
        float w = W[(r * DIM + i) * DIM + o];
        aq += w * q[r * DIM + o];
        ak += w * k[r * DIM + o];
    }
    qW[idx] = aq; kW[idx] = ak;
}

// ---------------- rel = relu(rel_bert @ rel_lin_w + b) ----------------
__global__ __launch_bounds__(256) void k_relemb(const float* __restrict__ bert, const float* __restrict__ w,
                                                const float* __restrict__ b, float* rel) {
    int idx = blockIdx.x * 256 + threadIdx.x;
    if (idx >= NR * DIM) return;
    int r = idx >> 8, o = idx & 255;
    float a = b[o];
    for (int i = 0; i < RELIN; i++) a += bert[r * RELIN + i] * w[i * DIM + o];
    rel[idx] = fmaxf(a, 0.f);
}

// ---------------- dq[n,r] = x[n]·qW[r], dk[n,r] = x[n]·kW[r] ----------------
__global__ __launch_bounds__(256) void k_dqdk(const float* __restrict__ x, const float* __restrict__ qW,
                                              const float* __restrict__ kW, float* dq, float* dk) {
    int idx = blockIdx.x * 256 + threadIdx.x;
    if (idx >= NN * NR) return;
    int r = idx & 31, n = idx >> 5;         // lanes share node n -> broadcast x reads
    float aq = 0.f, ak = 0.f;
    for (int i = 0; i < DIM; i++) {
        float xv = x[n * DIM + i];
        aq += xv * qW[r * DIM + i];
        ak += xv * kW[r * DIM + i];
    }
    dq[idx] = aq; dk[idx] = ak;
}

// ---------------- edge softmax passes ----------------
__global__ __launch_bounds__(256) void k_logit_max(const int* __restrict__ src, const int* __restrict__ dst,
                                                   const int* __restrict__ et, const float* __restrict__ dq,
                                                   const float* __restrict__ dk, float* logit, float* mbuf) {
    int e = blockIdx.x * 256 + threadIdx.x;
    if (e >= NE) return;
    int d = dst[e], s = src[e], r = et[e];
    float lg = dq[d * NR + r] + dk[s * NR + r];
    lg = lg > 0.f ? lg : 0.2f * lg;          // leaky_relu 0.2
    logit[e] = lg;
    int iv = __float_as_int(lg);
    if (iv >= 0) atomicMax((int*)mbuf + d, iv);
    else         atomicMin((unsigned int*)mbuf + d, (unsigned int)iv);
}

__global__ __launch_bounds__(256) void k_expsum(const int* __restrict__ dst, const float* __restrict__ mbuf,
                                                float* logit, float* sbuf) {
    int e = blockIdx.x * 256 + threadIdx.x;
    if (e >= NE) return;
    int d = dst[e];
    float w = __expf(logit[e] - mbuf[d]);
    logit[e] = w;
    atomicAdd(&sbuf[d], w);
}

__global__ __launch_bounds__(256) void k_norm(const int* __restrict__ dst, const float* __restrict__ sbuf,
                                              const float* __restrict__ wv, float* att) {
    int e = blockIdx.x * 256 + threadIdx.x;
    if (e >= NE) return;
    att[e] = wv[e] / (sbuf[dst[e]] + 1e-16f);
}

// ---------------- WMMA aggregation: out[dst] += a_e * (x[src] @ W[et]) ----------------
__global__ __launch_bounds__(256) void k_agg(const float* __restrict__ xIn, const unsigned short* __restrict__ Wp,
                                             const float* __restrict__ att, const int* __restrict__ srcS,
                                             const int* __restrict__ dstS, const int* __restrict__ eord,
                                             const int2* __restrict__ desc, float* out) {
    __shared__ unsigned short ldsA[16 * DIM];
    __shared__ int   ldsDst[16];
    __shared__ int   ldsSrc[16];
    __shared__ float ldsAtt[16];

    int2 d = desc[blockIdx.x];
    if (d.x < 0) return;                     // block-uniform: EXEC stays all-ones for WMMA
    int start = d.x, r = d.y & 0xFF, cnt = d.y >> 8;
    int tid = threadIdx.x;

    if (tid < 16) {
        if (tid < cnt) {
            ldsSrc[tid] = srcS[start + tid];
            ldsDst[tid] = dstS[start + tid];
            ldsAtt[tid] = att[eord[start + tid]];
        } else {
            ldsSrc[tid] = 0; ldsDst[tid] = 0; ldsAtt[tid] = 0.f;   // zero rows
        }
    }
    __syncthreads();

    // stage A = a_e * x[src] as bf16 [16][256]
    {
        int row = tid >> 4, c16 = (tid & 15) * 16;
        float a = ldsAtt[row];
        const float* xp = xIn + ldsSrc[row] * DIM + c16;
        #pragma unroll
        for (int j = 0; j < 16; j++) ldsA[row * DIM + c16 + j] = f2bf(xp[j] * a);
    }
    __syncthreads();

    int lane = tid & 31, wv = tid >> 5;
    int mb = (lane >> 4) << 3;               // 8 for upper lane half
    #pragma unroll
    for (int sub = 0; sub < 2; sub++) {
        int nt = wv * 2 + sub;
        v8f acc = {0.f,0.f,0.f,0.f,0.f,0.f,0.f,0.f};
        #pragma unroll
        for (int kk = 0; kk < 8; kk++) {
            const U4* pa = (const U4*)&ldsA[(lane & 15) * DIM + kk * 32 + mb];
            U8 ab; ab.lo = pa[0]; ab.hi = pa[2];            // K runs [0..7] and [16..23] (+halfshift)
            v16bf va = __builtin_bit_cast(v16bf, ab);
            const U4* pb = (const U4*)&Wp[(size_t)(((r * 8 + kk) * 16 + nt) * 32 + lane) * 16];
            U8 bb; bb.lo = pb[0]; bb.hi = pb[1];
            v16bf vb = __builtin_bit_cast(v16bf, bb);
            acc = __builtin_amdgcn_wmma_f32_16x16x32_bf16(false, va, false, vb, (short)0, acc, false, false);
        }
        int col = nt * 16 + (lane & 15);
        #pragma unroll
        for (int v = 0; v < 8; v++) {
            int M = v + mb;
            atomicAdd(&out[(size_t)ldsDst[M] * DIM + col], acc[v]);
        }
    }
}

__global__ __launch_bounds__(256) void k_relu(float* p, int n) {
    int i = blockIdx.x * 256 + threadIdx.x;
    if (i < n) p[i] = fmaxf(p[i], 0.f);
}

// ---------------- dense WMMA GEMM: hout = h2 @ lin_w + b ----------------
__global__ __launch_bounds__(256) void k_dense(const float* __restrict__ hIn, const unsigned short* __restrict__ Lp,
                                               const float* __restrict__ bias, float* out) {
    __shared__ unsigned short ldsA[16 * DIM];
    int tid = threadIdx.x;
    int n0 = blockIdx.x * 16;                // N = 10000 = 625 * 16 exact
    {
        int row = tid >> 4, c16 = (tid & 15) * 16;
        const float* xp = hIn + (size_t)(n0 + row) * DIM + c16;
        #pragma unroll
        for (int j = 0; j < 16; j++) ldsA[row * DIM + c16 + j] = f2bf(xp[j]);
    }
    __syncthreads();
    int lane = tid & 31, wv = tid >> 5;
    int mb = (lane >> 4) << 3;
    #pragma unroll
    for (int sub = 0; sub < 2; sub++) {
        int nt = wv * 2 + sub;
        v8f acc = {0.f,0.f,0.f,0.f,0.f,0.f,0.f,0.f};
        #pragma unroll
        for (int kk = 0; kk < 8; kk++) {
            const U4* pa = (const U4*)&ldsA[(lane & 15) * DIM + kk * 32 + mb];
            U8 ab; ab.lo = pa[0]; ab.hi = pa[2];
            v16bf va = __builtin_bit_cast(v16bf, ab);
            const U4* pb = (const U4*)&Lp[(size_t)((kk * 16 + nt) * 32 + lane) * 16];
            U8 bb; bb.lo = pb[0]; bb.hi = pb[1];
            v16bf vb = __builtin_bit_cast(v16bf, bb);
            acc = __builtin_amdgcn_wmma_f32_16x16x32_bf16(false, va, false, vb, (short)0, acc, false, false);
        }
        int col = nt * 16 + (lane & 15);
        #pragma unroll
        for (int v = 0; v < 8; v++)
            out[(size_t)(n0 + v + mb) * DIM + col] = acc[v] + bias[col];
    }
}

// ---------------- TransE scoring: one wave (32 lanes) per edge ----------------
__global__ __launch_bounds__(256) void k_score(const float* __restrict__ h, const float* __restrict__ rel,
                                               const int* __restrict__ src, const int* __restrict__ dst,
                                               const int* __restrict__ et, float* score) {
    int e = blockIdx.x * 8 + (threadIdx.x >> 5);   // NE divisible by 8
    int lane = threadIdx.x & 31;
    const float* ps = h   + (size_t)src[e] * DIM + lane * 8;
    const float* pd = h   + (size_t)dst[e] * DIM + lane * 8;
    const float* pr = rel + (size_t)et[e]  * DIM + lane * 8;
    float acc = 0.f;
    #pragma unroll
    for (int j = 0; j < 8; j++) acc += fabsf(ps[j] + pr[j] - pd[j]);
    #pragma unroll
    for (int o = 16; o > 0; o >>= 1) acc += __shfl_xor(acc, o, 32);
    if (lane == 0) score[e] = 10.0f - acc;
}

// =====================================================================
extern "C" void kernel_launch(void* const* d_in, const int* in_sizes, int n_in,
                              void* d_out, int out_size, void* d_ws, size_t ws_size,
                              hipStream_t stream) {
    const float* x     = (const float*)d_in[0];
    const int*   ei    = (const int*)  d_in[1];
    const int*   et    = (const int*)  d_in[2];
    const float* W1    = (const float*)d_in[3];
    const float* q1    = (const float*)d_in[4];
    const float* k1    = (const float*)d_in[5];
    const float* W2    = (const float*)d_in[6];
    const float* q2    = (const float*)d_in[7];
    const float* k2    = (const float*)d_in[8];
    const float* linw  = (const float*)d_in[9];
    const float* linb  = (const float*)d_in[10];
    const float* bert  = (const float*)d_in[11];
    const float* rlw   = (const float*)d_in[12];
    const float* rlb   = (const float*)d_in[13];
    const int* src = ei;
    const int* dst = ei + NE;
    float* score = (float*)d_out;

    // ---- workspace carve-up (~49 MB) ----
    char* wp = (char*)d_ws;
    auto alloc = [&](size_t b) { char* p = wp; wp += (b + 255) & ~(size_t)255; return p; };
    float* relE  = (float*)alloc((size_t)NR * DIM * 4);
    float* qW1   = (float*)alloc((size_t)NR * DIM * 4);
    float* kW1   = (float*)alloc((size_t)NR * DIM * 4);
    float* qW2   = (float*)alloc((size_t)NR * DIM * 4);
    float* kW2   = (float*)alloc((size_t)NR * DIM * 4);
    unsigned short* Wp1 = (unsigned short*)alloc((size_t)NR * 65536 * 2);
    unsigned short* Wp2 = (unsigned short*)alloc((size_t)NR * 65536 * 2);
    unsigned short* Lp  = (unsigned short*)alloc((size_t)65536 * 2);
    float* dq    = (float*)alloc((size_t)NN * NR * 4);
    float* dk    = (float*)alloc((size_t)NN * NR * 4);
    float* mbuf  = (float*)alloc((size_t)NN * 4);
    float* sbuf  = (float*)alloc((size_t)NN * 4);
    float* logit = (float*)alloc((size_t)NE * 4);
    float* att   = (float*)alloc((size_t)NE * 4);
    int*   srcS  = (int*)alloc((size_t)NE * 4);
    int*   dstS  = (int*)alloc((size_t)NE * 4);
    int*   eordS = (int*)alloc((size_t)NE * 4);
    int*   ibuf  = (int*)alloc(1024);                 // cnt[32] off[33] cur[32] toff[33]
    int2*  desc  = (int2*)alloc((size_t)MAX_TILES * sizeof(int2));
    float* h1    = (float*)alloc((size_t)NN * DIM * 4);
    float* h2    = (float*)alloc((size_t)NN * DIM * 4);
    float* hout  = (float*)alloc((size_t)NN * DIM * 4);

    const int EB = (NE + 255) / 256;                  // 1250
    const int DESCB = (MAX_TILES + 255) / 256;        // 79

    // ---- bucket edges by relation (launch-local, deterministic work) ----
    k_fill_u32<<<1, 256, 0, stream>>>((unsigned*)ibuf, 32, 0u);
    k_hist<<<EB, 256, 0, stream>>>(et, ibuf);
    k_scan<<<1, 1, 0, stream>>>(ibuf);
    k_desc<<<DESCB, 256, 0, stream>>>(ibuf, desc);
    k_sort<<<EB, 256, 0, stream>>>(src, dst, et, ibuf + 65, srcS, dstS, eordS);

    // ---- pack weights to WMMA B-fragment layout; small precomputes ----
    k_pack<<<(NR * 65536) / 256, 256, 0, stream>>>(W1, Wp1, NR);
    k_pack<<<(NR * 65536) / 256, 256, 0, stream>>>(W2, Wp2, NR);
    k_pack<<<65536 / 256, 256, 0, stream>>>(linw, Lp, 1);
    k_prep_qk<<<(NR * DIM) / 256, 256, 0, stream>>>(W1, q1, k1, qW1, kW1);
    k_prep_qk<<<(NR * DIM) / 256, 256, 0, stream>>>(W2, q2, k2, qW2, kW2);
    k_relemb<<<(NR * DIM) / 256, 256, 0, stream>>>(bert, rlw, rlb, relE);

    // ---- one RGAT layer ----
    auto layer = [&](const float* xin, const unsigned short* Wpk,
                     const float* qW, const float* kW, float* hOut) {
        k_dqdk<<<(NN * NR) / 256, 256, 0, stream>>>(xin, qW, kW, dq, dk);
        k_fill_u32<<<(NN + 255) / 256, 256, 0, stream>>>((unsigned*)mbuf, NN, 0xFF800000u); // -inf
        k_fill_u32<<<(NN + 255) / 256, 256, 0, stream>>>((unsigned*)sbuf, NN, 0u);
        k_logit_max<<<EB, 256, 0, stream>>>(src, dst, et, dq, dk, logit, mbuf);
        k_expsum<<<EB, 256, 0, stream>>>(dst, mbuf, logit, sbuf);
        k_norm<<<EB, 256, 0, stream>>>(dst, sbuf, logit, att);
        k_fill_u32<<<(NN * DIM) / 256, 256, 0, stream>>>((unsigned*)hOut, NN * DIM, 0u);
        k_agg<<<MAX_TILES, 256, 0, stream>>>(xin, Wpk, att, srcS, dstS, eordS, desc, hOut);
        k_relu<<<(NN * DIM) / 256, 256, 0, stream>>>(hOut, NN * DIM);
    };

    layer(x,  Wp1, qW1, kW1, h1);
    layer(h1, Wp2, qW2, kW2, h2);

    // ---- final linear + TransE score ----
    k_dense<<<NN / 16, 256, 0, stream>>>(h2, Lp, linb, hout);
    k_score<<<NE / 8, 256, 0, stream>>>(hout, relE, src, dst, et, score);

    (void)in_sizes; (void)n_in; (void)out_size; (void)ws_size;
}
